// MultiHeadLocalAttention_70377334112918
// MI455X (gfx1250) — compile-verified, matrix operand order checked by reading, not run on hardware
//
#include <hip/hip_runtime.h>
#include <hip/hip_bf16.h>

typedef __attribute__((ext_vector_type(16))) __bf16 v16bf;
typedef __attribute__((ext_vector_type(8)))  float  v8f;

#define D_MODEL 512
#define SEQ     2048
#define HEADS   8
#define DK      64

// ---------------------------------------------------------------------------
// fp32 -> bf16 conversion (4 elements / thread)
// ---------------------------------------------------------------------------
__global__ void cvt_f32_bf16(const float* __restrict__ in, __bf16* __restrict__ out, int n4) {
    int i = blockIdx.x * blockDim.x + threadIdx.x;
    if (i < n4) {
        float4 v = ((const float4*)in)[i];
        out[i * 4 + 0] = (__bf16)v.x;
        out[i * 4 + 1] = (__bf16)v.y;
        out[i * 4 + 2] = (__bf16)v.z;
        out[i * 4 + 3] = (__bf16)v.w;
    }
}

// ---------------------------------------------------------------------------
// W[k][n] (fp32, rows x cols) -> Wt[n][k] (bf16)  via LDS tile transpose
// ---------------------------------------------------------------------------
__global__ void transpose_cvt(const float* __restrict__ W, __bf16* __restrict__ Wt,
                              int rows, int cols) {
    __shared__ float tile[32][33];
    int bx = blockIdx.x * 32;            // col (n) base
    int by = blockIdx.y * 32;            // row (k) base
    int tx = threadIdx.x;                // 0..31
    int ty = threadIdx.y;                // 0..7
#pragma unroll
    for (int i = 0; i < 32; i += 8)
        tile[ty + i][tx] = W[(size_t)(by + ty + i) * cols + bx + tx];
    __syncthreads();
#pragma unroll
    for (int i = 0; i < 32; i += 8)
        Wt[(size_t)(bx + ty + i) * rows + by + tx] = (__bf16)tile[tx][ty + i];
}

// ---------------------------------------------------------------------------
// D = A(bf16 MxK) * Bt^T (Bt is NxK bf16) + bias  ->  fp32 MxN
//
// Block = 4 waves (128 thr) = 128(M) x 64(N) tile; each wave owns 32(M)x64(N).
// The 64-row Bt strip is staged in LDS (two 256-wide K halves, 64x264 bf16,
// +8 pad => 4-bank row skew, conflict-free ds_load_b128 per 16-lane half).
// Per k-step/wave: 2 global A fragments + 4 LDS B fragments loaded up front,
// then 8 back-to-back WMMAs (B reused across both M sub-tiles) so ds_loads
// overlap the matrix pipe instead of serializing on s_wait_dscnt.
// Fragment k order chosen here (same for A and B): lane&15 = row/col, lane>>4
// picks K 0..15 vs 16..31 -> contiguous 32B per lane.
// ---------------------------------------------------------------------------
#define KH   256                         // K elements staged per half
#define LDSW (KH + 8)                    // padded LDS row stride (elements)

union frag_u { uint4 u[2]; v16bf v; };

__global__ void gemm_bf16_wmma(const __bf16* __restrict__ A, const __bf16* __restrict__ Bt,
                               const float* __restrict__ bias, float* __restrict__ Dm,
                               int Mi, int Ni, int Ki) {
    __shared__ __bf16 Bs[64][LDSW];      // 33,792 bytes

    int tid  = threadIdx.x;
    int wid  = tid >> 5;                 // 0..3
    int lane = tid & 31;
    int l16  = lane & 15;
    int hl   = lane >> 4;                // 0: K 0..15, 1: K 16..31 of the step
    int m0   = blockIdx.y * 128 + wid * 32;
    int n0   = blockIdx.x * 64;

    v8f acc[2][4] = {};

    const __bf16* aRow0 = A + (size_t)(m0 + l16) * Ki + hl * 16;
    const __bf16* aRow1 = aRow0 + (size_t)16 * Ki;

    for (int kb = 0; kb < Ki; kb += KH) {
        __syncthreads();                 // previous half fully consumed
        // cooperative stage: 64 rows x 32 chunks(16B) = 2048 chunks / 128 thr
#pragma unroll
        for (int it = 0; it < 16; ++it) {
            int c   = it * 128 + tid;
            int row = c >> 5;
            int ch  = c & 31;
            uint4 val = *(const uint4*)(Bt + (size_t)(n0 + row) * Ki + kb + ch * 8);
            *(uint4*)(&Bs[row][ch * 8]) = val;
        }
        __syncthreads();

        for (int k0 = 0; k0 < KH; k0 += 32) {
            v16bf a0 = *(const v16bf*)(aRow0 + kb + k0);
            v16bf a1 = *(const v16bf*)(aRow1 + kb + k0);
            frag_u b[4];
#pragma unroll
            for (int t = 0; t < 4; ++t) {
                const __bf16* bp = &Bs[t * 16 + l16][k0 + hl * 16];
                b[t].u[0] = *(const uint4*)(bp);
                b[t].u[1] = *(const uint4*)(bp + 8);
            }
#pragma unroll
            for (int t = 0; t < 4; ++t)
                acc[0][t] = __builtin_amdgcn_wmma_f32_16x16x32_bf16(
                    false, a0, false, b[t].v, (short)0, acc[0][t], false, false);
#pragma unroll
            for (int t = 0; t < 4; ++t)
                acc[1][t] = __builtin_amdgcn_wmma_f32_16x16x32_bf16(
                    false, a1, false, b[t].v, (short)0, acc[1][t], false, false);
        }
    }

#pragma unroll
    for (int r = 0; r < 2; ++r) {
#pragma unroll
        for (int t = 0; t < 4; ++t) {
            int col = n0 + t * 16 + l16;
            float bv = bias[col];
#pragma unroll
            for (int e = 0; e < 8; ++e) {
                int row = m0 + r * 16 + hl * 8 + e;
                Dm[(size_t)row * Ni + col] = acc[r][t][e] + bv;
            }
        }
    }
}

// ---------------------------------------------------------------------------
// Vsum[b][c] = sum_s Vp[b][s][c]   (c = h*64+d), coalesced over c
// ---------------------------------------------------------------------------
__global__ void vsum_kernel(const float* __restrict__ Vp, float* __restrict__ Vsum,
                            int bs, int seq) {
    int t = blockIdx.x * blockDim.x + threadIdx.x;
    if (t >= bs * D_MODEL) return;
    int b = t / D_MODEL, c = t % D_MODEL;
    float acc = 0.f;
    for (int s = 0; s < seq; ++s)
        acc += Vp[((size_t)b * seq + s) * D_MODEL + c];
    Vsum[t] = acc;
}

// ---------------------------------------------------------------------------
// Banded count-weighted attention with full-row softmax background.
// C[a,b] = clamp(min+2, <=seq-1) - clamp(max-2, >=0) + 1  (zero when |a-b|>4).
// out = (sum_band (e_j - ebg) v_j + ebg * Vsum) / Z,
// Z = sum_band e_j + (seq - nband) * ebg,  ebg = exp(-m),  m = max(0, max s_j).
// One thread per (b, h, q); writes bf16 concat for the final WMMA GEMM.
// ---------------------------------------------------------------------------
__global__ void band_attention(const float* __restrict__ Qp, const float* __restrict__ Kp,
                               const float* __restrict__ Vp, const float* __restrict__ Vsum,
                               __bf16* __restrict__ concat, int bs, int seq) {
    int t = blockIdx.x * blockDim.x + threadIdx.x;
    if (t >= bs * HEADS * seq) return;
    int q = t % seq;
    int h = (t / seq) % HEADS;
    int b = t / (seq * HEADS);

    size_t rowQ = ((size_t)b * seq + q) * D_MODEL + h * DK;
    const float4* q4 = (const float4*)(Qp + rowQ);

    int lo = (q - 4 < 0) ? 0 : q - 4;
    int hi = (q + 4 > seq - 1) ? seq - 1 : q + 4;
    int nband = hi - lo + 1;

    float s[9];
    float m = 0.f;                         // background zeros participate in max
#pragma unroll
    for (int j = 0; j < 9; ++j) {
        int k = lo + j;
        bool valid = (k <= hi);
        int kk = valid ? k : hi;
        const float4* k4 = (const float4*)(Kp + ((size_t)b * seq + kk) * D_MODEL + h * DK);
        float dot = 0.f;
#pragma unroll
        for (int d = 0; d < 16; ++d) {
            float4 qa = q4[d], ka = k4[d];
            dot += qa.x * ka.x + qa.y * ka.y + qa.z * ka.z + qa.w * ka.w;
        }
        int mn = (q < kk) ? q : kk;
        int mx = (q < kk) ? kk : q;
        int cl = (mn + 2 < seq - 1) ? mn + 2 : seq - 1;
        int ch = (mx - 2 > 0) ? mx - 2 : 0;
        float C = (float)(cl - ch + 1);
        float sc = dot * C * 0.125f;       // 1/sqrt(64) = 0.125
        s[j] = valid ? sc : 0.f;
        if (valid) m = fmaxf(m, sc);
    }

    float ebg = __expf(-m);
    float Z = (float)(seq - nband) * ebg;
    float w[9];
#pragma unroll
    for (int j = 0; j < 9; ++j) {
        bool valid = (lo + j <= hi);
        float e = __expf(s[j] - m);
        Z += valid ? e : 0.f;
        w[j] = valid ? (e - ebg) : 0.f;
    }
    float inv = 1.f / Z;

    const float4* vs4 = (const float4*)(Vsum + (size_t)b * D_MODEL + h * DK);
    __bf16* outp = concat + rowQ;
#pragma unroll
    for (int d = 0; d < 16; ++d) {
        float4 vsv = vs4[d];
        float ax = ebg * vsv.x, ay = ebg * vsv.y, az = ebg * vsv.z, aw = ebg * vsv.w;
#pragma unroll
        for (int j = 0; j < 9; ++j) {
            int kk = (lo + j <= hi) ? lo + j : hi;   // w[j]==0 masks duplicates
            const float4* v4 = (const float4*)(Vp + ((size_t)b * seq + kk) * D_MODEL + h * DK);
            float4 vv = v4[d];
            ax += w[j] * vv.x; ay += w[j] * vv.y; az += w[j] * vv.z; aw += w[j] * vv.w;
        }
        outp[d * 4 + 0] = (__bf16)(ax * inv);
        outp[d * 4 + 1] = (__bf16)(ay * inv);
        outp[d * 4 + 2] = (__bf16)(az * inv);
        outp[d * 4 + 3] = (__bf16)(aw * inv);
    }
}

// ---------------------------------------------------------------------------
extern "C" void kernel_launch(void* const* d_in, const int* in_sizes, int n_in,
                              void* d_out, int out_size, void* d_ws, size_t ws_size,
                              hipStream_t stream) {
    const float* q  = (const float*)d_in[0];
    const float* k  = (const float*)d_in[1];
    const float* v  = (const float*)d_in[2];
    const float* Wq = (const float*)d_in[3];
    const float* bq = (const float*)d_in[4];
    const float* Wk = (const float*)d_in[5];
    const float* bk = (const float*)d_in[6];
    const float* Wv = (const float*)d_in[7];
    const float* bv = (const float*)d_in[8];
    const float* Wo = (const float*)d_in[9];
    const float* bo = (const float*)d_in[10];
    float* out = (float*)d_out;

    const int D = D_MODEL;
    const int M = in_sizes[0] / D;        // bs * seq
    const int BS = M / SEQ;

    // ---- workspace carve-out ----
    char* ws = (char*)d_ws;
    size_t off = 0;
    auto carve = [&](size_t bytes) -> void* {
        void* p = ws + off;
        off += (bytes + 255) & ~(size_t)255;
        return p;
    };
    __bf16* qa = (__bf16*)carve((size_t)M * D * 2);
    __bf16* ka = (__bf16*)carve((size_t)M * D * 2);
    __bf16* va = (__bf16*)carve((size_t)M * D * 2);
    __bf16* WqT = (__bf16*)carve((size_t)D * D * 2);
    __bf16* WkT = (__bf16*)carve((size_t)D * D * 2);
    __bf16* WvT = (__bf16*)carve((size_t)D * D * 2);
    __bf16* WoT = (__bf16*)carve((size_t)D * D * 2);
    float*  Qp  = (float*)carve((size_t)M * D * 4);
    float*  Kp  = (float*)carve((size_t)M * D * 4);
    float*  Vp  = (float*)carve((size_t)M * D * 4);
    float*  Vs  = (float*)carve((size_t)BS * D * 4);
    __bf16* cc  = (__bf16*)carve((size_t)M * D * 2);

    // 1) activations -> bf16
    int n4 = (M * D) / 4;
    int cblk = (n4 + 255) / 256;
    cvt_f32_bf16<<<cblk, 256, 0, stream>>>(q, qa, n4);
    cvt_f32_bf16<<<cblk, 256, 0, stream>>>(k, ka, n4);
    cvt_f32_bf16<<<cblk, 256, 0, stream>>>(v, va, n4);

    // 2) weights -> transposed bf16 [N][K]
    dim3 tb(32, 8), tg(D / 32, D / 32);
    transpose_cvt<<<tg, tb, 0, stream>>>(Wq, WqT, D, D);
    transpose_cvt<<<tg, tb, 0, stream>>>(Wk, WkT, D, D);
    transpose_cvt<<<tg, tb, 0, stream>>>(Wv, WvT, D, D);
    transpose_cvt<<<tg, tb, 0, stream>>>(Wo, WoT, D, D);

    // 3) projections via WMMA: [M,512] x [512,512] + bias
    dim3 gg(D / 64, M / 128);             // 8 x 32 blocks, 128 thr (4 waves)
    gemm_bf16_wmma<<<gg, 128, 0, stream>>>(qa, WqT, bq, Qp, M, D, D);
    gemm_bf16_wmma<<<gg, 128, 0, stream>>>(ka, WkT, bk, Kp, M, D, D);
    gemm_bf16_wmma<<<gg, 128, 0, stream>>>(va, WvT, bv, Vp, M, D, D);

    // 4) V column sums (softmax background term)
    vsum_kernel<<<(BS * D + 255) / 256, 256, 0, stream>>>(Vp, Vs, BS, SEQ);

    // 5) banded attention -> bf16 concat
    int at = BS * HEADS * SEQ;
    band_attention<<<(at + 255) / 256, 256, 0, stream>>>(Qp, Kp, Vp, Vs, cc, BS, SEQ);

    // 6) output projection -> d_out (fp32)
    gemm_bf16_wmma<<<gg, 128, 0, stream>>>(cc, WoT, bo, out, M, D, D);
}
